// SE3EquivariantLayer_16277926052601
// MI455X (gfx1250) — matrix operand order repaired and checked by reading, not compiled, not run on hardware
//
#include <hip/hip_runtime.h>
#include <hip/hip_bf16.h>
#include <math.h>

// ---- problem constants (B=2, N=512, D_IN=D_H=D_OUT=128) ----
#define NATOMS 512
#define DDIM   128
#define MROWS  1024   // B*N
#define PACK_ELEMS (DDIM * DDIM)   // 16384 bf16 per packed matrix

typedef __attribute__((ext_vector_type(16))) __bf16       v16bf;
typedef __attribute__((ext_vector_type(8)))  float        v8f;
typedef __attribute__((ext_vector_type(2)))  float        v2f;
typedef __attribute__((ext_vector_type(4)))  unsigned int v4u;
typedef __attribute__((ext_vector_type(8)))  int          v8i;
typedef __attribute__((ext_vector_type(4)))  int          v4i;

__device__ __forceinline__ float silu_f(float x) {
    return x / (1.0f + __expf(-x));
}

// =====================================================================
// Kernel P: pre-pack the five 128x128 f32 weight matrices into bf16,
// laid out in v_wmma_f32_16x16x32_bf16 B-fragment order so the GEMM
// kernels load one contiguous 32B v16bf per lane (no cvt in hot loop).
// Fragment f = (kchunk)*8 + ntile (kchunk=kb/32, ntile=n0/16);
// element (lane L, e): K = kb + 16*(e>>3) + 8*(L>=16) + (e&7),
//                      N = n0 + (L&15).
// =====================================================================
__global__ __launch_bounds__(256)
void pack_weights_kernel(const float* __restrict__ Ws1, const float* __restrict__ Ws2,
                         const float* __restrict__ Wv1, const float* __restrict__ Wv2,
                         const float* __restrict__ Wa1, __bf16* __restrict__ pack)
{
    const int idx = blockIdx.x * 256 + threadIdx.x;   // 0 .. 5*16384-1
    const int mat = idx >> 14;
    const int r   = idx & (PACK_ELEMS - 1);
    const int f   = r >> 9;            // fragment 0..31
    const int L   = (r >> 4) & 31;     // lane
    const int e   = r & 15;            // element within lane
    const int kb  = (f >> 3) << 5;
    const int n0  = (f & 7) << 4;
    const int K   = kb + ((e >> 3) << 4) + ((L >> 4) << 3) + (e & 7);
    const int N   = n0 + (L & 15);
    const float* W = (mat == 0) ? Ws1 : (mat == 1) ? Ws2 :
                     (mat == 2) ? Wv1 : (mat == 3) ? Wv2 : Wa1;
    pack[idx] = (__bf16)W[K * DDIM + N];
}

// =====================================================================
// Kernel A: per-node GEMMs via v_wmma_f32_16x16x32_bf16.
//   scalar = SiLU(nf@Ws1+bs1)@Ws2+bs2            -> d_out[0 : 1024*128)
//   vf3    = (SiLU(nf@Wv1+bv1)@Wv2+bv2)[:, :3]   -> ws
//   pre    = nf@Wa1[:128] + ba1                  -> ws (pre-activation)
// Block = 16 output rows, 8 waves; wave w owns col tile n0=16w.
// Input tile DMA'd into LDS (f32) by the TDM (wave 0 issues a 2D D#,
// s_wait_tensorcnt 0, barrier), then converted ONCE to a bf16 LDS copy
// so the stage-1 A-fragment loads are pure ds_load_b128.
// =====================================================================
__global__ __launch_bounds__(256)
void node_mlps_kernel(const float* __restrict__ nf,
                      const float* __restrict__ bs1, const float* __restrict__ bs2,
                      const float* __restrict__ bv1, const float* __restrict__ bv2,
                      const float* __restrict__ ba1,
                      const __bf16* __restrict__ pack,   // 5 packed matrices
                      float* __restrict__ scalar_out,    // [1024,128]
                      float* __restrict__ pre,           // [1024,128] ws
                      float* __restrict__ vf3)           // [1024,3]   ws
{
    __shared__ __bf16 nfa[16][DDIM];  // bf16 input tile (A-fragment source)
    __shared__ __bf16 hs [16][DDIM];  // SiLU hidden, scalar branch
    __shared__ __bf16 hv [16][DDIM];  // SiLU hidden, vector branch

    const int tid  = threadIdx.x;
    const int lane = tid & 31;
    const int wave = tid >> 5;
    const int rowbase = blockIdx.x * 16;

    const __bf16* pWs1 = pack + 0 * PACK_ELEMS;
    const __bf16* pWs2 = pack + 1 * PACK_ELEMS;
    const __bf16* pWv1 = pack + 2 * PACK_ELEMS;
    const __bf16* pWv2 = pack + 3 * PACK_ELEMS;
    const __bf16* pWa1 = pack + 4 * PACK_ELEMS;

#if __has_builtin(__builtin_amdgcn_tensor_load_to_lds)
    __shared__ float nfs_f32[16][DDIM];   // TDM destination (raw f32)
    // ---- TDM: DMA the 16x128 f32 tile global -> LDS (2D descriptor) ----
    if (wave == 0) {
        const unsigned ldsOff = (unsigned)(size_t)(&nfs_f32[0][0]);    // LDS byte offset (low 32 bits)
        const unsigned long long ga =
            (unsigned long long)(size_t)(nf + (size_t)rowbase * DDIM);
        v4u g0;
        g0[0] = 1u;                                   // count=1 (valid user D#)
        g0[1] = ldsOff;                               // lds_addr
        g0[2] = (unsigned)(ga & 0xFFFFFFFFu);         // global_addr[31:0]
        g0[3] = (unsigned)((ga >> 32) & 0x01FFFFFFu)  // global_addr[56:32]
              | (2u << 30);                           // type=2 ("image")
        v8i g1;
        g1[0] = (int)(2u << 16);      // data_size=2 (4 bytes/elem)
        g1[1] = (int)(128u << 16);    // tensor_dim0 = 128 (lo16 @ bits 63:48)
        g1[2] = (int)(16u  << 16);    // tensor_dim1 = 16
        g1[3] = (int)(128u << 16);    // tile_dim0 = 128
        g1[4] = 16;                   // tile_dim1 = 16, tile_dim2 = 0
        g1[5] = 128;                  // tensor_dim0_stride = 128
        g1[6] = 0;
        g1[7] = 0;
        v4i gz = {0, 0, 0, 0};
#if __clang_major__ >= 23
        v8i gz8 = {0, 0, 0, 0, 0, 0, 0, 0};
        __builtin_amdgcn_tensor_load_to_lds(g0, g1, gz, gz, gz8, 0);
#else
        __builtin_amdgcn_tensor_load_to_lds(g0, g1, gz, gz, 0);
#endif
        __builtin_amdgcn_s_wait_tensorcnt(0);
    }
    __syncthreads();
    // one-time f32 -> bf16 conversion (4 pk-cvts per thread)
    #pragma unroll
    for (int e = 0; e < 8; ++e) {
        int idx = tid * 8 + e;
        int r = idx >> 7, c = idx & 127;
        nfa[r][c] = (__bf16)nfs_f32[r][c];
    }
    __syncthreads();
#else
    #pragma unroll
    for (int e = 0; e < 8; ++e) {
        int idx = tid * 8 + e;
        int r = idx >> 7, c = idx & 127;
        nfa[r][c] = (__bf16)nf[(rowbase + r) * DDIM + c];
    }
    __syncthreads();
#endif

    const int nn   = wave * 16 + (lane & 15);   // this lane's output column
    const int mh   = (lane >> 4) << 3;          // +8 row offset, upper half
    const int kA8  = (lane >> 4) << 3;          // A: +8 K offset, upper half
    const int m    = lane & 15;

    // ---- stage 1: three GEMMs share the A tile; bias pre-loaded into C ----
    v8f cs, cv, ca;
    {
        float b1 = bs1[nn], b2 = bv1[nn], b3 = ba1[nn];
        #pragma unroll
        for (int r = 0; r < 8; ++r) { cs[r] = b1; cv[r] = b2; ca[r] = b3; }
    }
    for (int kb = 0; kb < DDIM; kb += 32) {
        const int fragOff = ((((kb >> 5) * 8) + wave) * 32 + lane) * 16;
        if (kb + 32 < DDIM)   // prefetch next K-chunk fragment, keep near (WGP scope)
            __builtin_prefetch(pWs1 + fragOff + 8 * 32 * 16, 0, 3);
        v16bf a;
        #pragma unroll
        for (int v = 0; v < 8; ++v) {
            int k = kb + ((v & 3) << 1) + ((v >> 2) << 4) + kA8;
            a[2*v]   = nfa[m][k];
            a[2*v+1] = nfa[m][k+1];
        }
        v16bf bS = *(const v16bf*)(pWs1 + fragOff);
        v16bf bV = *(const v16bf*)(pWv1 + fragOff);
        v16bf bA = *(const v16bf*)(pWa1 + fragOff);
        cs = __builtin_amdgcn_wmma_f32_16x16x32_bf16(false, a, false, bS, (short)0, cs, false, false);
        cv = __builtin_amdgcn_wmma_f32_16x16x32_bf16(false, a, false, bV, (short)0, cv, false, false);
        ca = __builtin_amdgcn_wmma_f32_16x16x32_bf16(false, a, false, bA, (short)0, ca, false, false);
    }
    #pragma unroll
    for (int r = 0; r < 8; ++r) {
        int mm = r + mh;
        hs[mm][nn] = (__bf16)silu_f(cs[r]);
        hv[mm][nn] = (__bf16)silu_f(cv[r]);
        pre[(rowbase + mm) * DDIM + nn] = ca[r];   // pre-activation (no SiLU)
    }
    __syncthreads();

    // ---- stage 2: second Linear of both MLPs ----
    v8f c2s, c2v;
    {
        float b1 = bs2[nn], b2 = bv2[nn];
        #pragma unroll
        for (int r = 0; r < 8; ++r) { c2s[r] = b1; c2v[r] = b2; }
    }
    for (int kb = 0; kb < DDIM; kb += 32) {
        const int fragOff = ((((kb >> 5) * 8) + wave) * 32 + lane) * 16;
        v16bf aS, aV;
        #pragma unroll
        for (int v = 0; v < 8; ++v) {
            int k = kb + ((v & 3) << 1) + ((v >> 2) << 4) + kA8;
            aS[2*v]   = hs[m][k];
            aS[2*v+1] = hs[m][k+1];
            aV[2*v]   = hv[m][k];
            aV[2*v+1] = hv[m][k+1];
        }
        v16bf bS = *(const v16bf*)(pWs2 + fragOff);
        v16bf bV = *(const v16bf*)(pWv2 + fragOff);
        // all waves run both WMMAs (EXEC all-ones); only wave 0 stores vf3
        c2s = __builtin_amdgcn_wmma_f32_16x16x32_bf16(false, aS, false, bS, (short)0, c2s, false, false);
        c2v = __builtin_amdgcn_wmma_f32_16x16x32_bf16(false, aV, false, bV, (short)0, c2v, false, false);
    }
    #pragma unroll
    for (int r = 0; r < 8; ++r) {
        int mm = r + mh;
        scalar_out[(rowbase + mm) * DDIM + nn] = c2s[r];
        if (wave == 0 && nn < 3)
            vf3[(rowbase + mm) * 3 + nn] = c2v[r];
    }
}

// =====================================================================
// Kernel B: attention via rank-3 update on v_wmma_f32_16x16x4_f32, then
// masked softmax + direction-weighted aggregation. One block per (b,i);
// wave w owns j in [64w, 64w+64), 4 chunks of 16. All per-block operands
// (pre_i, Wa2, Wa1 rows 128..130, coords) staged in LDS once.
// =====================================================================
__global__ __launch_bounds__(256)
void attention_kernel(const float* __restrict__ coords,  // [2,512,3]
                      const int*   __restrict__ emask,   // [2,512,512]
                      const float* __restrict__ Wa1,     // [131,128]
                      const float* __restrict__ Wa2,     // [128,1]
                      const float* __restrict__ ba2,     // [1]
                      const float* __restrict__ pre,     // [1024,128] ws
                      const float* __restrict__ vf3,     // [1024,3]   ws
                      float* __restrict__ vec_out)       // [1024,3]
{
    __shared__ float lg[NATOMS];        // logits, then exp()
    __shared__ float red[256];
    __shared__ float preS[DDIM], wa2S[DDIM];
    __shared__ float w0S[DDIM], w1S[DDIM], w2S[DDIM];
    __shared__ float xyzS[NATOMS][3];   // this batch's coordinates

    const int tid  = threadIdx.x;
    const int lane = tid & 31;
    const int wave = tid >> 5;
    const int bi   = blockIdx.x;     // 0..1023 == b*512 + i
    const int b    = bi >> 9;
    const int i    = bi & 511;

    if (tid < DDIM) {
        preS[tid] = pre[bi * DDIM + tid];
        wa2S[tid] = Wa2[tid];
        w0S[tid]  = Wa1[128 * DDIM + tid];
        w1S[tid]  = Wa1[129 * DDIM + tid];
        w2S[tid]  = Wa1[130 * DDIM + tid];
    }
    #pragma unroll
    for (int e = 0; e < 6; ++e) {      // 512*3 = 1536 floats, 6 per thread
        int idx = tid * 6 + e;
        xyzS[idx / 3][idx % 3] = coords[(b * NATOMS) * 3 + idx];
    }
    __syncthreads();

    const float xi0 = xyzS[i][0], xi1 = xyzS[i][1], xi2 = xyzS[i][2];
    const float bias2 = ba2[0];
    const bool  hi = (lane >= 16);

    for (int chunk = 0; chunk < 4; ++chunk) {
        const int jbase = wave * 64 + chunk * 16;
        const int jm = jbase + (lane & 15);       // row this lane feeds

        // distance features for row jm (safe_norm semantics)
        const float dx = xi0 - xyzS[jm][0];
        const float dy = xi1 - xyzS[jm][1];
        const float dz = xi2 - xyzS[jm][2];
        const float d2 = dx*dx + dy*dy + dz*dz;
        const float dist = (d2 > 0.0f) ? sqrtf(d2) : 0.0f;
        const float f0 = dist;
        const float f1 = 1.0f / (dist + 1e-6f);
        const float f2 = __expf(-dist);

        // A 16x4 f32: lanes 0-15 carry K=(0,1)=(d, 1/d); lanes 16-31 K=(2,3)=(e^-d, 0)
        v2f a;
        a[0] = hi ? f2 : f0;
        a[1] = hi ? 0.0f : f1;

        float s[8];
        #pragma unroll
        for (int r = 0; r < 8; ++r) s[r] = 0.0f;

        for (int t = 0; t < 8; ++t) {            // 8 column tiles cover D=128
            const int n = t * 16 + (lane & 15);
            v2f bm;                              // B 4x16: Wa1 rows 128..130, zero pad
            bm[0] = hi ? w2S[n] : w0S[n];
            bm[1] = hi ? 0.0f   : w1S[n];
            const float pv = preS[n];
            v8f c;
            #pragma unroll
            for (int r = 0; r < 8; ++r) c[r] = pv;   // C[m][n] = pre[i][n]
            c = __builtin_amdgcn_wmma_f32_16x16x4_f32(false, a, false, bm, (short)0, c, false, false);
            const float w2 = wa2S[n];
            #pragma unroll
            for (int r = 0; r < 8; ++r)
                s[r] += silu_f(c[r]) * w2;       // partial of h . Wa2
        }
        // reduce the 128-dot across the 16 lanes of each half-wave
        #pragma unroll
        for (int r = 0; r < 8; ++r) {
            s[r] += __shfl_xor(s[r], 1, 16);
            s[r] += __shfl_xor(s[r], 2, 16);
            s[r] += __shfl_xor(s[r], 4, 16);
            s[r] += __shfl_xor(s[r], 8, 16);
        }
        // lanes 0-15 hold rows 0..7; lanes 16-31 hold rows 8..15
        if (lane == 0 || lane == 16) {
            const int rbase = (lane >> 4) << 3;
            #pragma unroll
            for (int r = 0; r < 8; ++r) {
                const int j = jbase + rbase + r;
                float v = s[r] + bias2;
                if (emask[(b * NATOMS + i) * NATOMS + j] == 0) v = -__builtin_inff();
                lg[j] = v;
            }
        }
    }
    __syncthreads();

    // ---- masked softmax over j (512 entries, 256 threads) ----
    red[tid] = fmaxf(lg[tid], lg[tid + 256]);
    __syncthreads();
    for (int off = 128; off > 0; off >>= 1) {
        if (tid < off) red[tid] = fmaxf(red[tid], red[tid + off]);
        __syncthreads();
    }
    const float mx = red[0];
    __syncthreads();

    float e0 = __expf(lg[tid]       - mx);
    float e1 = __expf(lg[tid + 256] - mx);
    lg[tid]       = e0;            // own slots only; no race
    lg[tid + 256] = e1;
    red[tid] = e0 + e1;
    __syncthreads();
    for (int off = 128; off > 0; off >>= 1) {
        if (tid < off) red[tid] += red[tid + off];
        __syncthreads();
    }
    const float inv = 1.0f / red[0];
    __syncthreads();

    // ---- direction-weighted aggregation: sum_j w_j * (x_i - x_j) ----
    float p0 = 0.0f, p1 = 0.0f, p2 = 0.0f;
    for (int jj = tid; jj < NATOMS; jj += 256) {
        const float w = lg[jj] * inv;
        p0 += w * (xi0 - xyzS[jj][0]);
        p1 += w * (xi1 - xyzS[jj][1]);
        p2 += w * (xi2 - xyzS[jj][2]);
    }
    float agg[3];
    float pc[3] = {p0, p1, p2};
    #pragma unroll
    for (int c3 = 0; c3 < 3; ++c3) {
        red[tid] = pc[c3];
        __syncthreads();
        for (int off = 128; off > 0; off >>= 1) {
            if (tid < off) red[tid] += red[tid + off];
            __syncthreads();
        }
        agg[c3] = red[0];
        __syncthreads();
    }
    if (tid == 0) {
        vec_out[bi*3 + 0] = agg[0] * vf3[bi*3 + 0];
        vec_out[bi*3 + 1] = agg[1] * vf3[bi*3 + 1];
        vec_out[bi*3 + 2] = agg[2] * vf3[bi*3 + 2];
    }
}

// =====================================================================
extern "C" void kernel_launch(void* const* d_in, const int* in_sizes, int n_in,
                              void* d_out, int out_size, void* d_ws, size_t ws_size,
                              hipStream_t stream) {
    const float* nf  = (const float*)d_in[0];
    const float* xyz = (const float*)d_in[1];
    const int*   em  = (const int*)  d_in[2];
    const float* Ws1 = (const float*)d_in[3];
    const float* bs1 = (const float*)d_in[4];
    const float* Ws2 = (const float*)d_in[5];
    const float* bs2 = (const float*)d_in[6];
    const float* Wv1 = (const float*)d_in[7];
    const float* bv1 = (const float*)d_in[8];
    const float* Wv2 = (const float*)d_in[9];
    const float* bv2 = (const float*)d_in[10];
    const float* Wa1 = (const float*)d_in[11];
    const float* ba1 = (const float*)d_in[12];
    const float* Wa2 = (const float*)d_in[13];
    const float* ba2 = (const float*)d_in[14];

    float* scalar_out = (float*)d_out;                    // [1024,128]
    float* vec_out    = (float*)d_out + MROWS * DDIM;     // [1024,3]

    float*  pre  = (float*)d_ws;                          // [1024,128]
    float*  vf3  = pre + MROWS * DDIM;                    // [1024,3]
    __bf16* pack = (__bf16*)(vf3 + MROWS * 3);            // 5 * 16384 bf16 (32B-aligned)

    pack_weights_kernel<<<5 * PACK_ELEMS / 256, 256, 0, stream>>>(
        Ws1, Ws2, Wv1, Wv2, Wa1, pack);

    node_mlps_kernel<<<MROWS / 16, 256, 0, stream>>>(
        nf, bs1, bs2, bv1, bv2, ba1, pack, scalar_out, pre, vf3);

    attention_kernel<<<MROWS, 256, 0, stream>>>(
        xyz, em, Wa1, Wa2, ba2, pre, vf3, vec_out);
}
